// simsgl_frame_84731114816076
// MI455X (gfx1250) — compile-verified, matrix operand order checked by reading, not run on hardware
//
#include <hip/hip_runtime.h>

// ---------------- problem constants ----------------
#define N_USERS   100000
#define N_ITEMS   50000
#define N_NODES   150000
#define NNZ       2400000
#define DIM       64
#define BATCH     4096
#define EPS_F     0.1f
#define CL_RATE_F 0.2f
#define REG_F     1e-4f
#define INV_TEMP  5.0f          // 1 / 0.2

typedef __attribute__((ext_vector_type(16))) _Float16 v16h;
typedef __attribute__((ext_vector_type(8)))  float    v8f;

// ---------------- small device helpers ----------------
__device__ __forceinline__ float hash_uniform(unsigned a, unsigned b, unsigned c) {
    unsigned h = a * 0x9E3779B1u ^ b * 0x85EBCA77u ^ c * 0xC2B2AE3Du;
    h ^= h >> 16; h *= 0x7FEB352Du;
    h ^= h >> 15; h *= 0x846CA68Bu;
    h ^= h >> 16;
    return (float)(h >> 8) * (1.0f / 16777216.0f);   // [0,1)
}

__device__ __forceinline__ float sgnf(float x) {
    return (x > 0.0f) ? 1.0f : ((x < 0.0f) ? -1.0f : 0.0f);
}

// ---------------- SpMM: y[rows[e]] += vals[e] * x[cols[e]] ----------------
// wave32 per nnz; lane handles 2 of the 64 dims (coalesced float2).
// nnz metadata is wave-uniform: readfirstlane -> scalar s_load path.
__global__ __launch_bounds__(256) void spmm_kernel(
    const float* __restrict__ x, float* __restrict__ y,
    const int* __restrict__ rows, const int* __restrict__ cols,
    const float* __restrict__ vals)
{
    int wid  = (blockIdx.x * blockDim.x + threadIdx.x) >> 5;
    if (wid >= NNZ) return;
    int lane = threadIdx.x & 31;

    int eu = __builtin_amdgcn_readfirstlane(wid);   // provably wave-uniform index
    int   r = rows[eu];
    int   c = cols[eu];
    float v = vals[eu];

    // prefetch the gather row a few nnz ahead (global_prefetch_b8)
    int nxt = eu + 8; if (nxt >= NNZ) nxt = NNZ - 1;
    int c2 = cols[nxt];
    __builtin_prefetch(x + (size_t)c2 * DIM, 0, 0);

    const float2 xv = *(const float2*)(x + (size_t)c * DIM + lane * 2);
    float* yp = y + (size_t)r * DIM + lane * 2;
    atomicAdd(yp,     v * xv.x);
    atomicAdd(yp + 1, v * xv.y);
}

// ---------------- per-layer: optional noise + mean accumulation ----------------
__global__ __launch_bounds__(256) void noise_acc_kernel(
    float* __restrict__ cur, float* __restrict__ acc,
    int use_noise, unsigned seed)
{
    int wid  = (blockIdx.x * blockDim.x + threadIdx.x) >> 5;
    if (wid >= N_NODES) return;
    int lane = threadIdx.x & 31;

    float* p = cur + (size_t)wid * DIM + lane * 2;
    float2 f = *(float2*)p;

    if (use_noise) {
        float n0 = hash_uniform(seed, (unsigned)wid, (unsigned)(lane * 2));
        float n1 = hash_uniform(seed, (unsigned)wid, (unsigned)(lane * 2 + 1));
        float ssq = n0 * n0 + n1 * n1;
        #pragma unroll
        for (int m = 16; m >= 1; m >>= 1) ssq += __shfl_xor(ssq, m, 32);
        float sc = EPS_F * rsqrtf(ssq + 1e-20f);
        f.x += sgnf(f.x) * n0 * sc;
        f.y += sgnf(f.y) * n1 * sc;
        *(float2*)p = f;
    }

    float* ap = acc + (size_t)wid * DIM + lane * 2;
    ap[0] += f.x;
    ap[1] += f.y;
}

// ---------------- BPR rec loss + Frobenius reg sums ----------------
__global__ __launch_bounds__(256) void rec_loss_kernel(
    const float* __restrict__ acc,
    const int* __restrict__ ul, const int* __restrict__ pl, const int* __restrict__ nl,
    float* __restrict__ sc)
{
    int wid  = (blockIdx.x * blockDim.x + threadIdx.x) >> 5;
    if (wid >= BATCH) return;
    int lane = threadIdx.x & 31;

    int u = ul[wid], p = pl[wid], n = nl[wid];
    const float2 ue = *(const float2*)(acc + (size_t)u * DIM + lane * 2);
    const float2 pe = *(const float2*)(acc + (size_t)(N_USERS + p) * DIM + lane * 2);
    const float2 ne = *(const float2*)(acc + (size_t)(N_USERS + n) * DIM + lane * 2);

    float dp = ue.x * pe.x + ue.y * pe.y;
    float dn = ue.x * ne.x + ue.y * ne.y;
    float su = ue.x * ue.x + ue.y * ue.y;
    float sp = pe.x * pe.x + pe.y * pe.y;
    #pragma unroll
    for (int m = 16; m >= 1; m >>= 1) {
        dp += __shfl_xor(dp, m, 32);
        dn += __shfl_xor(dn, m, 32);
        su += __shfl_xor(su, m, 32);
        sp += __shfl_xor(sp, m, 32);
    }
    if (lane == 0) {
        const float inv9 = 1.0f / 9.0f;
        float x   = (dp - dn) * inv9;
        float sig = 1.0f / (1.0f + __expf(-x));
        atomicAdd(&sc[0], -logf(1e-7f + sig));
        atomicAdd(&sc[1], su);
        atomicAdd(&sc[2], sp);
    }
}

// ---------------- unique: mark + compact ----------------
__global__ void mark_kernel(const int* __restrict__ ul, const int* __restrict__ pl,
                            int* __restrict__ fu, int* __restrict__ fi)
{
    int t = blockIdx.x * blockDim.x + threadIdx.x;
    if (t >= BATCH) return;
    fu[ul[t]] = 1;
    fi[pl[t]] = 1;
}

__global__ void compact_kernel(const int* __restrict__ flags, int n,
                               int* __restrict__ idx, int* __restrict__ cnt)
{
    int t = blockIdx.x * blockDim.x + threadIdx.x;
    if (t >= n) return;
    if (flags[t]) {
        int pos = atomicAdd(cnt, 1);
        idx[pos] = t;
    }
}

// ---------------- gather + L2-normalize + fp16 pack ----------------
__global__ __launch_bounds__(256) void norm_gather_kernel(
    const float* __restrict__ acc, const int* __restrict__ idx,
    const int* __restrict__ cntp, int base, _Float16* __restrict__ out)
{
    int wid = (blockIdx.x * blockDim.x + threadIdx.x) >> 5;
    if (wid >= *cntp) return;
    int lane = threadIdx.x & 31;

    int row = base + idx[wid];
    const float2 f = *(const float2*)(acc + (size_t)row * DIM + lane * 2);
    float ssq = f.x * f.x + f.y * f.y;
    #pragma unroll
    for (int m = 16; m >= 1; m >>= 1) ssq += __shfl_xor(ssq, m, 32);
    float inv = rsqrtf(ssq + 1e-20f);

    _Float16* op = out + (size_t)wid * DIM + lane * 2;
    op[0] = (_Float16)(f.x * inv);
    op[1] = (_Float16)(f.y * inv);
}

// ---------------- InfoNCE denominator via WMMA, streaming logsumexp ----------------
// 4 waves/block, 64-row strip. Each 16-col B tile (2KB) is staged into LDS ONCE
// per block via async global->LDS (128 threads x b128), fenced with
// s_wait_asynccnt, then shared by all 4 waves' chained WMMAs (K=64 -> 2x
// v_wmma_f32_16x16x32_f16). Normalized dots are in [-1,1] -> logits in [-5,5]:
// exp is safe without max-shift, so logsumexp streams with O(1) state per row.
__global__ __launch_bounds__(128) void infonce_kernel(
    const _Float16* __restrict__ A, const _Float16* __restrict__ B,
    const int* __restrict__ cntp, float* __restrict__ ttl_sum)
{
    __shared__ _Float16 btile[16 * DIM];    // 2KB: one 16-col tile, all K=64

    int cnt = *cntp;
    int blk_row = blockIdx.x * 64;
    if (blk_row >= cnt) return;             // block-uniform: whole block skips barriers

    int tid  = threadIdx.x;
    int wave = tid >> 5;
    int lane = tid & 31;
    int hsel = lane >> 4;                   // 0: lanes 0-15, 1: lanes 16-31
    int lid  = lane & 15;
    int row_base = blk_row + wave * 16;     // this wave's 16-row strip

    // A fragments (16x32 f16 each); fixed rows, reused across all column tiles.
    const _Float16* arow = A + (size_t)(row_base + lid) * DIM + hsel * 16;
    v16h a0 = *(const v16h*)(arow);         // K =  0..31
    v16h a1 = *(const v16h*)(arow + 32);    // K = 32..63

    float accum[8];
    #pragma unroll
    for (int v = 0; v < 8; ++v) accum[v] = 0.0f;

    // each thread owns one 16B slice of the staged tile
    unsigned lds_dst = (unsigned)(uintptr_t)(&btile[tid * 8]);

    int ntiles = (cnt + 15) >> 4;
    for (int t = 0; t < ntiles; ++t) {
        int col_base = t * 16;

        // async stage: 128 threads x 16B = 2KB (rows col_base..col_base+15, K=0..63)
        {
            const _Float16* src = B + (size_t)col_base * DIM + tid * 8;
            unsigned long long ga = (unsigned long long)(uintptr_t)src;
            asm volatile("global_load_async_to_lds_b128 %0, %1, off"
                         :: "v"(lds_dst), "v"(ga) : "memory");
            asm volatile("s_wait_asynccnt 0" ::: "memory");
        }
        __syncthreads();

        // B fragment from LDS: B[k][n] = v2[col_base+n][k] -> contiguous halves
        const _Float16* brow = &btile[lid * DIM + hsel * 16];
        v16h b0 = *(const v16h*)(brow);
        v16h b1 = *(const v16h*)(brow + 32);

        v8f c = {};
        c = __builtin_amdgcn_wmma_f32_16x16x32_f16(false, a0, false, b0,
                                                   (short)0, c, false, false);
        c = __builtin_amdgcn_wmma_f32_16x16x32_f16(false, a1, false, b1,
                                                   (short)0, c, false, false);

        bool colOk = (col_base + lid) < cnt;
        #pragma unroll
        for (int v = 0; v < 8; ++v) {
            float e = __expf(c[v] * INV_TEMP);
            accum[v] += colOk ? e : 0.0f;
        }
        __syncthreads();                    // before next overwrite of btile
    }

    // C layout: lane half h, VGPR v -> row (v + 8*h), column lid.
    #pragma unroll
    for (int v = 0; v < 8; ++v) {
        float s = accum[v];
        #pragma unroll
        for (int m = 8; m >= 1; m >>= 1) s += __shfl_xor(s, m, 16);
        if (lid == 0) {
            int row = row_base + v + 8 * hsel;
            if (row < cnt) atomicAdd(ttl_sum, logf(s));
        }
    }
}

// ---------------- positive-pair dots (diagonal of logits) ----------------
__global__ __launch_bounds__(256) void posdot_kernel(
    const _Float16* __restrict__ A, const _Float16* __restrict__ B,
    const int* __restrict__ cntp, float* __restrict__ pos_sum)
{
    int wid = (blockIdx.x * blockDim.x + threadIdx.x) >> 5;
    if (wid >= *cntp) return;
    int lane = threadIdx.x & 31;

    const _Float16* ap = A + (size_t)wid * DIM + lane * 2;
    const _Float16* bp = B + (size_t)wid * DIM + lane * 2;
    float d = (float)ap[0] * (float)bp[0] + (float)ap[1] * (float)bp[1];
    #pragma unroll
    for (int m = 16; m >= 1; m >>= 1) d += __shfl_xor(d, m, 32);
    if (lane == 0) atomicAdd(pos_sum, d * INV_TEMP);
}

// ---------------- final combine ----------------
__global__ void final_kernel(const float* __restrict__ sc,
                             const int* __restrict__ cnts,
                             float* __restrict__ out)
{
    float rec = sc[0] / (float)BATCH;
    float reg = REG_F * (sqrtf(sc[1]) * (1.0f / 3.0f) + sqrtf(sc[2]) * (1.0f / 3.0f));
    int cu = cnts[0] > 0 ? cnts[0] : 1;
    int ci = cnts[1] > 0 ? cnts[1] : 1;
    float cl_u = (sc[3] - sc[4]) / (float)cu;
    float cl_i = (sc[5] - sc[6]) / (float)ci;
    float cl = CL_RATE_F * (cl_u + cl_i);
    out[0] = rec + reg + cl;
    out[1] = cl;
}

// ---------------- host orchestration ----------------
extern "C" void kernel_launch(void* const* d_in, const int* in_sizes, int n_in,
                              void* d_out, int out_size, void* d_ws, size_t ws_size,
                              hipStream_t stream) {
    const float* user_embed = (const float*)d_in[0];
    const float* item_embed = (const float*)d_in[1];
    const float* adj_vals   = (const float*)d_in[2];
    const int*   adj_rows   = (const int*)d_in[3];
    const int*   adj_cols   = (const int*)d_in[4];
    const int*   user_list  = (const int*)d_in[5];
    const int*   pos_list   = (const int*)d_in[6];
    const int*   neg_list   = (const int*)d_in[7];
    float* out = (float*)d_out;

    // ---- workspace layout (~118 MB) ----
    const size_t NODE_F = (size_t)N_NODES * DIM;             // floats per node buffer
    float*    buf0  = (float*)d_ws;
    float*    buf1  = buf0 + NODE_F;
    float*    acc   = buf1 + NODE_F;
    _Float16* a1u   = (_Float16*)(acc + NODE_F);
    _Float16* a2u   = a1u + (size_t)BATCH * DIM;
    _Float16* a1i   = a2u + (size_t)BATCH * DIM;
    _Float16* a2i   = a1i + (size_t)BATCH * DIM;
    int*      fu    = (int*)(a2i + (size_t)BATCH * DIM);
    int*      fi    = fu + N_USERS;
    int*      u_idx = fi + N_ITEMS;
    int*      i_idx = u_idx + BATCH;
    int*      cnts  = i_idx + BATCH;          // [0]=cnt_u [1]=cnt_i
    float*    sc    = (float*)(cnts + 2);     // [0]=rec [1]=ssq_u [2]=ssq_p
                                              // [3]=ttl_u [4]=pos_u [5]=ttl_i [6]=pos_i

    const size_t NODE_BYTES = NODE_F * sizeof(float);
    const int SPMM_BLOCKS  = (NNZ + 7) / 8;          // 8 waves/block
    const int NODE_BLOCKS  = (N_NODES + 7) / 8;
    const int BATCH_BLOCKS = (BATCH + 7) / 8;

    // ---- zero-init mutable state (harness does not re-poison between replays) ----
    hipMemsetAsync(fu,   0, (size_t)(N_USERS + N_ITEMS) * sizeof(int), stream);
    hipMemsetAsync(cnts, 0, 2 * sizeof(int) + 7 * sizeof(float), stream);
    hipMemsetAsync(a1u,  0, 4 * (size_t)BATCH * DIM * sizeof(_Float16), stream);

    // ---- unique sets (depend only on inputs) ----
    mark_kernel<<<(BATCH + 255) / 256, 256, 0, stream>>>(user_list, pos_list, fu, fi);
    compact_kernel<<<(N_USERS + 255) / 256, 256, 0, stream>>>(fu, N_USERS, u_idx, &cnts[0]);
    compact_kernel<<<(N_ITEMS + 255) / 256, 256, 0, stream>>>(fi, N_ITEMS, i_idx, &cnts[1]);

    // ---- 3-layer LightGCN encode (optionally perturbed), acc = sum of layers ----
    auto encode = [&](int use_noise, unsigned seed) {
        hipMemcpyAsync(buf0, user_embed, (size_t)N_USERS * DIM * sizeof(float),
                       hipMemcpyDeviceToDevice, stream);
        hipMemcpyAsync(buf0 + (size_t)N_USERS * DIM, item_embed,
                       (size_t)N_ITEMS * DIM * sizeof(float),
                       hipMemcpyDeviceToDevice, stream);
        hipMemsetAsync(acc, 0, NODE_BYTES, stream);
        float* in  = buf0;
        float* outb = buf1;
        for (int k = 0; k < 3; ++k) {
            hipMemsetAsync(outb, 0, NODE_BYTES, stream);
            spmm_kernel<<<SPMM_BLOCKS, 256, 0, stream>>>(in, outb, adj_rows, adj_cols, adj_vals);
            noise_acc_kernel<<<NODE_BLOCKS, 256, 0, stream>>>(outb, acc, use_noise,
                                                              seed * 0x1000193u + (unsigned)k);
            float* t = in; in = outb; outb = t;
        }
    };

    // main (unperturbed) view -> rec loss + reg sums
    encode(0, 0u);
    rec_loss_kernel<<<BATCH_BLOCKS, 256, 0, stream>>>(acc, user_list, pos_list, neg_list, sc);

    // perturbed view 1
    encode(1, 101u);
    norm_gather_kernel<<<BATCH_BLOCKS, 256, 0, stream>>>(acc, u_idx, &cnts[0], 0,       a1u);
    norm_gather_kernel<<<BATCH_BLOCKS, 256, 0, stream>>>(acc, i_idx, &cnts[1], N_USERS, a1i);

    // perturbed view 2
    encode(1, 202u);
    norm_gather_kernel<<<BATCH_BLOCKS, 256, 0, stream>>>(acc, u_idx, &cnts[0], 0,       a2u);
    norm_gather_kernel<<<BATCH_BLOCKS, 256, 0, stream>>>(acc, i_idx, &cnts[1], N_USERS, a2i);

    // InfoNCE: WMMA-tiled streaming logsumexp + diagonal positives
    infonce_kernel<<<BATCH / 64, 128, 0, stream>>>(a1u, a2u, &cnts[0], &sc[3]);
    posdot_kernel<<<BATCH_BLOCKS, 256, 0, stream>>>(a1u, a2u, &cnts[0], &sc[4]);
    infonce_kernel<<<BATCH / 64, 128, 0, stream>>>(a1i, a2i, &cnts[1], &sc[5]);
    posdot_kernel<<<BATCH_BLOCKS, 256, 0, stream>>>(a1i, a2i, &cnts[1], &sc[6]);

    final_kernel<<<1, 1, 0, stream>>>(sc, cnts, out);
}
